// MultiFusionVoxel_56349970923787
// MI455X (gfx1250) — matrix operand order repaired and checked by reading, not compiled, not run on hardware
//
#include <hip/hip_runtime.h>
#include <hip/hip_bf16.h>
#include <stdint.h>

// ---------------------------------------------------------------------------
// Multi-scale VFE for MI455X (gfx1250, wave32).
//  - dense-grid occupancy + prefix scan  == jnp.unique (sorted) semantics
//  - layer1 (K=10) scalar FMA (memory bound, K too small for WMMA)
//  - layer2 GEMM: bf16 WMMA 16x16x32, f32 accum; run twice (BN stats pass +
//    apply pass) instead of storing the 268MB pre-BN tensor.
// ---------------------------------------------------------------------------

#define NPTS 262144
#define TGX 176
#define TGY 200
#define TGZ 1

typedef float  v8f   __attribute__((ext_vector_type(8)));
typedef __bf16 v16bf __attribute__((ext_vector_type(16)));
typedef __bf16 v8bf  __attribute__((ext_vector_type(8)));

__device__ __forceinline__ __bf16 f2bf(float f) {
    unsigned u = __builtin_bit_cast(unsigned, f);
    unsigned r = (u + 0x7FFFu + ((u >> 16) & 1u)) >> 16;   // RNE
    unsigned short h = (unsigned short)r;
    return __builtin_bit_cast(__bf16, h);
}

__device__ __forceinline__ int clampi(int v, int lo, int hi) {
    return v < lo ? lo : (v > hi ? hi : v);
}

// ---------------------------------------------------------------- zero fill
__global__ void k_zero(uint4* p, size_t n16) {
    size_t i = (size_t)blockIdx.x * blockDim.x + threadIdx.x;
    size_t s = (size_t)gridDim.x * blockDim.x;
    uint4 z = {0u, 0u, 0u, 0u};
    for (; i < n16; i += s) p[i] = z;
}

// ------------------------------------------------------------- voxel counts
__global__ void k_count(const float* __restrict__ pts, int n,
                        float vx, float vy, float vz, int gx, int gy, int gz,
                        int* __restrict__ cnt) {
    int i = blockIdx.x * blockDim.x + threadIdx.x;
    if (i >= n) return;
    const float* p = pts + (size_t)i * 5;
    int b  = (int)p[0];
    int cx = clampi((int)floorf(p[1] / vx), 0, gx - 1);
    int cy = clampi((int)floorf((p[2] + 40.f) / vy), 0, gy - 1);
    int cz = clampi((int)floorf((p[3] + 3.f) / vz), 0, gz - 1);
    int lin = ((b * gz + cz) * gy + cy) * gx + cx;
    atomicAdd(&cnt[lin], 1);
}

// ------------------------------------------- occupancy scan (3-level, exact)
__global__ void k_blocksum(const int* __restrict__ cnt, int cells,
                           int* __restrict__ bsum) {
    __shared__ int sh[256];
    int base = blockIdx.x * 1024;
    int s = 0;
    for (int j = 0; j < 4; j++) {
        int idx = base + threadIdx.x * 4 + j;
        if (idx < cells) s += (cnt[idx] > 0);
    }
    sh[threadIdx.x] = s;
    __syncthreads();
    for (int off = 128; off > 0; off >>= 1) {
        if (threadIdx.x < off) sh[threadIdx.x] += sh[threadIdx.x + off];
        __syncthreads();
    }
    if (threadIdx.x == 0) bsum[blockIdx.x] = sh[0];
}

__global__ void k_scan2(int* bs, int nb, int* total) {
    __shared__ int sh[256];
    __shared__ int carry;
    if (threadIdx.x == 0) carry = 0;
    __syncthreads();
    for (int base = 0; base < nb; base += 256) {
        int i = base + threadIdx.x;
        int v = (i < nb) ? bs[i] : 0;
        sh[threadIdx.x] = v;
        __syncthreads();
        for (int off = 1; off < 256; off <<= 1) {
            int t = (threadIdx.x >= off) ? sh[threadIdx.x - off] : 0;
            __syncthreads();
            sh[threadIdx.x] += t;
            __syncthreads();
        }
        int incl = sh[threadIdx.x];
        int c = carry;
        if (i < nb) bs[i] = c + incl - v;       // exclusive prefix
        __syncthreads();
        if (threadIdx.x == 255) carry = c + sh[255];
        __syncthreads();
    }
    if (threadIdx.x == 0) *total = carry;       // M = #occupied voxels
}

__global__ void k_scanfinal(const int* __restrict__ cnt, int cells,
                            const int* __restrict__ boff,
                            int* __restrict__ pref, int* __restrict__ vlin,
                            float* __restrict__ vcnt) {
    __shared__ int sh[256];
    int base = blockIdx.x * 1024 + threadIdx.x * 4;
    int f[4], c[4], ls = 0;
    for (int j = 0; j < 4; j++) {
        int idx = base + j;
        int v = (idx < cells) ? cnt[idx] : 0;
        c[j] = v; f[j] = (v > 0); ls += f[j];
    }
    sh[threadIdx.x] = ls;
    __syncthreads();
    for (int off = 1; off < 256; off <<= 1) {
        int t = (threadIdx.x >= off) ? sh[threadIdx.x - off] : 0;
        __syncthreads();
        sh[threadIdx.x] += t;
        __syncthreads();
    }
    int rank = boff[blockIdx.x] + sh[threadIdx.x] - ls;
    for (int j = 0; j < 4; j++) {
        int idx = base + j;
        if (idx < cells) {
            pref[idx] = rank;
            if (f[j]) { vlin[rank] = idx; vcnt[rank] = (float)c[j]; }
        }
        rank += f[j];
    }
}

// ----------------------------------------------- per-point inv + xyz segsum
__global__ void k_inv(const float* __restrict__ pts, int n,
                      float vx, float vy, float vz, int gx, int gy, int gz,
                      const int* __restrict__ pref, int* __restrict__ inv,
                      float* __restrict__ sumxyz) {
    int i = blockIdx.x * blockDim.x + threadIdx.x;
    if (i >= n) return;
    const float* p = pts + (size_t)i * 5;
    int b  = (int)p[0];
    int cx = clampi((int)floorf(p[1] / vx), 0, gx - 1);
    int cy = clampi((int)floorf((p[2] + 40.f) / vy), 0, gy - 1);
    int cz = clampi((int)floorf((p[3] + 3.f) / vz), 0, gz - 1);
    int lin = ((b * gz + cz) * gy + cy) * gx + cx;
    int r = pref[lin];
    inv[i] = r;
    atomicAdd(&sumxyz[3 * r + 0], p[1]);
    atomicAdd(&sumxyz[3 * r + 1], p[2]);
    atomicAdd(&sumxyz[3 * r + 2], p[3]);
}

__device__ __forceinline__ void build_feats(const float* p,
                                            float vx, float vy, float vz,
                                            int gx, int gy, int gz, int r,
                                            const float* sumxyz,
                                            const float* vcnt, float* fe) {
    int cx = clampi((int)floorf(p[1] / vx), 0, gx - 1);
    int cy = clampi((int)floorf((p[2] + 40.f) / vy), 0, gy - 1);
    int cz = clampi((int)floorf((p[3] + 3.f) / vz), 0, gz - 1);
    float ic = 1.f / vcnt[r];
    fe[0] = p[1]; fe[1] = p[2]; fe[2] = p[3]; fe[3] = p[4];
    fe[4] = p[1] - sumxyz[3 * r + 0] * ic;
    fe[5] = p[2] - sumxyz[3 * r + 1] * ic;
    fe[6] = p[3] - sumxyz[3 * r + 2] * ic;
    fe[7] = p[1] - ((cx + 0.5f) * vx);
    fe[8] = p[2] - (-40.f + (cy + 0.5f) * vy);
    fe[9] = p[3] - (-3.f + (cz + 0.5f) * vz);
}

// ------------------------------------------------ layer1: BN stats (pass 1)
template <int C1>
__global__ void k_l1_stats(const float* __restrict__ pts, int n,
                           float vx, float vy, float vz, int gx, int gy, int gz,
                           const int* __restrict__ inv,
                           const float* __restrict__ sumxyz,
                           const float* __restrict__ vcnt,
                           const float* __restrict__ W1,
                           float* __restrict__ bns, float* __restrict__ bnq) {
    __shared__ float w[10 * C1];
    __shared__ float ss[C1], sq[C1];
    for (int i = threadIdx.x; i < 10 * C1; i += blockDim.x) w[i] = W1[i];
    for (int i = threadIdx.x; i < C1; i += blockDim.x) { ss[i] = 0.f; sq[i] = 0.f; }
    __syncthreads();
    int i = blockIdx.x * blockDim.x + threadIdx.x;
    if (i < n) {
        const float* p = pts + (size_t)i * 5;
        float fe[10];
        build_feats(p, vx, vy, vz, gx, gy, gz, inv[i], sumxyz, vcnt, fe);
        for (int c = 0; c < C1; c++) {
            float y = 0.f;
#pragma unroll
            for (int k = 0; k < 10; k++) y = fmaf(fe[k], w[k * C1 + c], y);
            atomicAdd(&ss[c], y);          // ds_add_f32
            atomicAdd(&sq[c], y * y);
        }
    }
    __syncthreads();
    for (int c = threadIdx.x; c < C1; c += blockDim.x) {
        atomicAdd(&bns[c], ss[c]);
        atomicAdd(&bnq[c], sq[c]);
    }
}

// ----------------------------------------------------------- BN a/b params
__global__ void k_bnparams(const float* s, const float* q, const float* g,
                           const float* b, int C, float invN,
                           float* pa, float* pb) {
    int c = threadIdx.x;
    if (c >= C) return;
    float m = s[c] * invN;
    float v = q[c] * invN - m * m;
    float a = g[c] * rsqrtf(v + 1e-3f);
    pa[c] = a;
    pb[c] = b[c] - m * a;
}

// --------------------------------- layer1: apply BN+relu, write bf16, segsum
template <int C1>
__global__ void k_l1_apply(const float* __restrict__ pts, int n,
                           float vx, float vy, float vz, int gx, int gy, int gz,
                           const int* __restrict__ inv,
                           const float* __restrict__ sumxyz,
                           const float* __restrict__ vcnt,
                           const float* __restrict__ W1,
                           const float* __restrict__ pa,
                           const float* __restrict__ pb,
                           __bf16* __restrict__ hcat,
                           float* __restrict__ voxsum1) {
    __shared__ float w[10 * C1];
    for (int i = threadIdx.x; i < 10 * C1; i += blockDim.x) w[i] = W1[i];
    __syncthreads();
    int i = blockIdx.x * blockDim.x + threadIdx.x;
    if (i >= n) return;
    const float* p = pts + (size_t)i * 5;
    int r = inv[i];
    float fe[10];
    build_feats(p, vx, vy, vz, gx, gy, gz, r, sumxyz, vcnt, fe);
    for (int c = 0; c < C1; c++) {
        float y = 0.f;
#pragma unroll
        for (int k = 0; k < 10; k++) y = fmaf(fe[k], w[k * C1 + c], y);
        float h = fmaxf(0.f, pa[c] * y + pb[c]);
        hcat[(size_t)i * (2 * C1) + c] = f2bf(h);
        atomicAdd(&voxsum1[(size_t)r * C1 + c], h);
    }
}

// --------------------------- concat voxel-mean(h1) into hcat second half
template <int C1>
__global__ void k_voxmean(int n, const int* __restrict__ inv,
                          const float* __restrict__ vcnt,
                          const float* __restrict__ voxsum1,
                          __bf16* __restrict__ hcat) {
    int i = blockIdx.x * blockDim.x + threadIdx.x;
    if (i >= n) return;
    int r = inv[i];
    float ic = 1.f / vcnt[r];
    __bf16* dst = hcat + (size_t)i * (2 * C1) + C1;
    const float* src = voxsum1 + (size_t)r * C1;
    for (int c = 0; c < C1; c++) dst[c] = f2bf(src[c] * ic);
}

// ------------------------ pack W2 (f32) into per-lane WMMA B-fragment order
__global__ void k_w2pack(const float* __restrict__ W2, int K, int C2,
                         __bf16* __restrict__ pack) {
    int idx = blockIdx.x * blockDim.x + threadIdx.x;
    int total = C2 * K;                      // (C2/16)*(K/32)*32*16
    if (idx >= total) return;
    int e  = idx & 15;
    int L  = (idx >> 4) & 31;
    int nkt = K / 32;
    int kt = (idx >> 9) % nkt;
    int ct = idx / (512 * nkt);
    int kk  = kt * 32 + ((L >> 4) << 3) + (e & 7) + ((e >> 3) << 4); // lanes 0-15: K 0..7,16..23 ; 16-31: K 8..15,24..31
    int col = ct * 16 + (L & 15);
    pack[idx] = f2bf(W2[(size_t)kk * C2 + col]);
}

// ----------------------------------------------- GEMM2 fragment loader (A)
__device__ __forceinline__ v16bf load_afrag(const __bf16* arow, int kt, int base0) {
    v8bf lo = *(const v8bf*)(arow + kt * 32 + base0);
    v8bf hi = *(const v8bf*)(arow + kt * 32 + 16 + base0);
    v16bf a;
#pragma unroll
    for (int e = 0; e < 8; e++) { a[e] = lo[e]; a[8 + e] = hi[e]; }
    return a;
}

// ------------------------------------- GEMM2 pass A: BN statistics (WMMA)
template <int KDIM, int C2>
__global__ __launch_bounds__(32 * (C2 / 16))
void k_gemm_stats(const __bf16* __restrict__ hcat,
                  const __bf16* __restrict__ pack,
                  float* __restrict__ bns, float* __restrict__ bnq) {
    constexpr int NKT = KDIM / 32;
    __shared__ float ss[C2], sq[C2];
    for (int i = threadIdx.x; i < C2; i += blockDim.x) { ss[i] = 0.f; sq[i] = 0.f; }
    __syncthreads();
    int lane = threadIdx.x & 31;
    int ct   = threadIdx.x >> 5;
    int rowA = blockIdx.x * 16 + (lane & 15);
    int base0 = (lane >> 4) * 8;
    const __bf16* arow = hcat + (size_t)rowA * KDIM;
    __builtin_prefetch(arow + (size_t)16 * KDIM, 0, 1);   // next row-tile
    v8f acc = {};
#pragma unroll
    for (int kt = 0; kt < NKT; kt++) {
        v16bf a = load_afrag(arow, kt, base0);
        v16bf b = *(const v16bf*)(pack + (((size_t)ct * NKT + kt) * 32 + lane) * 16);
        acc = __builtin_amdgcn_wmma_f32_16x16x32_bf16(false, a, false, b,
                                                      (short)0, acc, false, false);
    }
    int col = ct * 16 + (lane & 15);
#pragma unroll
    for (int r = 0; r < 8; r++) {
        float v = acc[r];
        atomicAdd(&ss[col], v);
        atomicAdd(&sq[col], v * v);
    }
    __syncthreads();
    for (int i = threadIdx.x; i < C2; i += blockDim.x) {
        atomicAdd(&bns[i], ss[i]);
        atomicAdd(&bnq[i], sq[i]);
    }
}

// ------------------- GEMM2 pass B: recompute, BN+relu, voxel segsum (WMMA)
template <int KDIM, int C2>
__global__ __launch_bounds__(32 * (C2 / 16))
void k_gemm_apply(const __bf16* __restrict__ hcat,
                  const __bf16* __restrict__ pack,
                  const float* __restrict__ pa, const float* __restrict__ pb,
                  const int* __restrict__ inv,
                  float* __restrict__ voxsum2) {
    constexpr int NKT = KDIM / 32;
    int lane = threadIdx.x & 31;
    int ct   = threadIdx.x >> 5;
    int rowA = blockIdx.x * 16 + (lane & 15);
    int base0 = (lane >> 4) * 8;
    const __bf16* arow = hcat + (size_t)rowA * KDIM;
    __builtin_prefetch(arow + (size_t)16 * KDIM, 0, 1);
    v8f acc = {};
#pragma unroll
    for (int kt = 0; kt < NKT; kt++) {
        v16bf a = load_afrag(arow, kt, base0);
        v16bf b = *(const v16bf*)(pack + (((size_t)ct * NKT + kt) * 32 + lane) * 16);
        acc = __builtin_amdgcn_wmma_f32_16x16x32_bf16(false, a, false, b,
                                                      (short)0, acc, false, false);
    }
    int col = ct * 16 + (lane & 15);
    float a2 = pa[col], b2 = pb[col];
    int rbase = blockIdx.x * 16 + ((lane >> 4) << 3);
#pragma unroll
    for (int r = 0; r < 8; r++) {
        float h = fmaxf(0.f, a2 * acc[r] + b2);
        int iv = inv[rbase + r];
        atomicAdd(&voxsum2[(size_t)iv * C2 + col], h);
    }
}

// ----------------------------------------------------------- vf finalize
__global__ void k_vf(const float* __restrict__ vox, const float* __restrict__ vcnt,
                     const int* __restrict__ Mc, float* __restrict__ out,
                     int C2, int n) {
    int idx = blockIdx.x * blockDim.x + threadIdx.x;
    if (idx >= n * C2) return;
    int row = idx / C2;
    out[idx] = (row < *Mc) ? vox[idx] / vcnt[row] : 0.f;
}

// ------------------------------------------- med/low -> top merge (segsum)
__global__ void k_merge(const int* __restrict__ vlin, const float* __restrict__ vcnt,
                        const float* __restrict__ vox, const int* __restrict__ Mc,
                        int gx, int gy, int gz, int factor,
                        const int* __restrict__ vlinT, const int* __restrict__ McT,
                        float* __restrict__ msum, float* __restrict__ mcnt, int C2) {
    int j = blockIdx.x * blockDim.x + threadIdx.x;
    if (j >= *Mc) return;
    int lin = vlin[j];
    int x = lin % gx; int r = lin / gx;
    int y = r % gy;   r /= gy;
    int z = r % gz;   int b = r / gz;
    int parent = ((b * TGZ + z / factor) * TGY + y / factor) * TGX + x / factor;
    int MT = *McT;
    int lo = 0, hi = MT;
    while (lo < hi) { int mid = (lo + hi) >> 1; if (vlinT[mid] < parent) lo = mid + 1; else hi = mid; }
    if (lo >= MT || vlinT[lo] != parent) return;
    float ic = 1.f / vcnt[j];
    for (int c = 0; c < C2; c++)
        atomicAdd(&msum[(size_t)lo * C2 + c], vox[(size_t)j * C2 + c] * ic);
    atomicAdd(&mcnt[lo], 1.0f);
}

__global__ void k_mergedfinal(const float* __restrict__ msum, const float* __restrict__ mcnt,
                              const float* __restrict__ lsum, const float* __restrict__ lcnt,
                              const int* __restrict__ McT, float* __restrict__ out, int n) {
    int idx = blockIdx.x * blockDim.x + threadIdx.x;
    if (idx >= n * 128) return;
    int row = idx >> 7, c = idx & 127;
    float v = 0.f;
    if (row < *McT) {
        if (c < 64) v = msum[(size_t)row * 64 + c] / fmaxf(mcnt[row], 1.f);
        else        v = lsum[(size_t)row * 64 + (c - 64)] / fmaxf(lcnt[row], 1.f);
    }
    out[idx] = v;
}

__global__ void k_coors(const int* __restrict__ vlinT, const int* __restrict__ McT,
                        float* __restrict__ out, int n) {
    int row = blockIdx.x * blockDim.x + threadIdx.x;
    if (row >= n) return;
    float4 v;
    if (row < *McT) {
        int lin = vlinT[row];
        int x = lin % TGX; int r = lin / TGX;
        int y = r % TGY;   r /= TGY;
        int z = r % TGZ;   int b = r / TGZ;
        v = make_float4((float)b, (float)z, (float)y, (float)x);
    } else {
        v = make_float4(-1.f, -1.f, -1.f, -1.f);
    }
    ((float4*)out)[row] = v;
}

// ===========================================================================
struct Scale {
    float vx, vy, vz; int gx, gy, gz; int cells;
    int* count; int* pref; float* sumxyz; int* inv; int* vlin; float* vcnt;
    float *bn1s, *bn1q, *bn1a, *bn1b, *bn2s, *bn2q, *bn2a, *bn2b;
    float* voxsum1; float* voxsum2; __bf16* hcat; __bf16* pack;
    const float *W1, *g1, *b1, *W2, *g2, *b2;
    int* Mc;
};

template <int C1>
static void run_scale(const Scale& s, const float* pts, int* bsum, hipStream_t st) {
    constexpr int K2 = 2 * C1;               // GEMM2: K = C2 = 2*C1
    int nb = (NPTS + 255) / 256;
    k_count<<<nb, 256, 0, st>>>(pts, NPTS, s.vx, s.vy, s.vz, s.gx, s.gy, s.gz, s.count);
    int sb = (s.cells + 1023) / 1024;
    k_blocksum<<<sb, 256, 0, st>>>(s.count, s.cells, bsum);
    k_scan2<<<1, 256, 0, st>>>(bsum, sb, s.Mc);
    k_scanfinal<<<sb, 256, 0, st>>>(s.count, s.cells, bsum, s.pref, s.vlin, s.vcnt);
    k_inv<<<nb, 256, 0, st>>>(pts, NPTS, s.vx, s.vy, s.vz, s.gx, s.gy, s.gz,
                              s.pref, s.inv, s.sumxyz);
    k_l1_stats<C1><<<nb, 256, 0, st>>>(pts, NPTS, s.vx, s.vy, s.vz, s.gx, s.gy, s.gz,
                                       s.inv, s.sumxyz, s.vcnt, s.W1, s.bn1s, s.bn1q);
    k_bnparams<<<1, C1, 0, st>>>(s.bn1s, s.bn1q, s.g1, s.b1, C1, 1.f / NPTS, s.bn1a, s.bn1b);
    k_l1_apply<C1><<<nb, 256, 0, st>>>(pts, NPTS, s.vx, s.vy, s.vz, s.gx, s.gy, s.gz,
                                       s.inv, s.sumxyz, s.vcnt, s.W1, s.bn1a, s.bn1b,
                                       s.hcat, s.voxsum1);
    k_voxmean<C1><<<nb, 256, 0, st>>>(NPTS, s.inv, s.vcnt, s.voxsum1, s.hcat);
    int packTotal = K2 * K2;
    k_w2pack<<<(packTotal + 255) / 256, 256, 0, st>>>(s.W2, K2, K2, s.pack);
    k_gemm_stats<K2, K2><<<NPTS / 16, 32 * (K2 / 16), 0, st>>>(s.hcat, s.pack, s.bn2s, s.bn2q);
    k_bnparams<<<1, K2, 0, st>>>(s.bn2s, s.bn2q, s.g2, s.b2, K2, 1.f / NPTS, s.bn2a, s.bn2b);
    k_gemm_apply<K2, K2><<<NPTS / 16, 32 * (K2 / 16), 0, st>>>(s.hcat, s.pack,
                                                               s.bn2a, s.bn2b, s.inv, s.voxsum2);
}

extern "C" void kernel_launch(void* const* d_in, const int* in_sizes, int n_in,
                              void* d_out, int out_size, void* d_ws, size_t ws_size,
                              hipStream_t stream) {
    const float* pts = (const float*)d_in[0];
    const int CELLS_T = 2 * TGZ * TGY * TGX;       // 70400
    const int CELLS_M = 2 * 2 * 400 * 352;         // 563200
    const int CELLS_L = 2 * 4 * 800 * 704;         // 4505600
    const int MMAX = NPTS;

    char* cur = (char*)d_ws;
    auto take = [&](size_t bytes) -> char* {
        char* p = cur;
        cur += (bytes + 255) & ~(size_t)255;
        return p;
    };

    // ----- zero-initialized region (atomic accumulators) -----
    char* zbase = cur;
    int*   countT = (int*)take((size_t)CELLS_T * 4);
    int*   countM = (int*)take((size_t)CELLS_M * 4);
    int*   countL = (int*)take((size_t)CELLS_L * 4);
    float* sumxyzT = (float*)take((size_t)CELLS_T * 3 * 4);
    float* sumxyzM = (float*)take((size_t)MMAX * 3 * 4);
    float* sumxyzL = (float*)take((size_t)MMAX * 3 * 4);
    float* voxsum1T = (float*)take((size_t)CELLS_T * 64 * 4);
    float* voxsum1M = (float*)take((size_t)MMAX * 32 * 4);
    float* voxsum1L = (float*)take((size_t)MMAX * 32 * 4);
    float* voxsum2T = (float*)take((size_t)CELLS_T * 128 * 4);
    float* voxsum2M = (float*)take((size_t)MMAX * 64 * 4);
    float* voxsum2L = (float*)take((size_t)MMAX * 64 * 4);
    float* msum = (float*)take((size_t)CELLS_T * 64 * 4);
    float* mcnt = (float*)take((size_t)CELLS_T * 4);
    float* lsum = (float*)take((size_t)CELLS_T * 64 * 4);
    float* lcnt = (float*)take((size_t)CELLS_T * 4);
    float* bn1sT = (float*)take(64 * 4); float* bn1qT = (float*)take(64 * 4);
    float* bn1sM = (float*)take(32 * 4); float* bn1qM = (float*)take(32 * 4);
    float* bn1sL = (float*)take(32 * 4); float* bn1qL = (float*)take(32 * 4);
    float* bn2sT = (float*)take(128 * 4); float* bn2qT = (float*)take(128 * 4);
    float* bn2sM = (float*)take(64 * 4);  float* bn2qM = (float*)take(64 * 4);
    float* bn2sL = (float*)take(64 * 4);  float* bn2qL = (float*)take(64 * 4);
    size_t zbytes = (size_t)(cur - zbase);

    // ----- non-zeroed scratch -----
    int* prefT = (int*)take((size_t)CELLS_T * 4);
    int* prefM = (int*)take((size_t)CELLS_M * 4);
    int* prefL = (int*)take((size_t)CELLS_L * 4);
    int* bsum  = (int*)take(4608 * 4);
    int* Mc    = (int*)take(3 * 4);
    int* invT  = (int*)take((size_t)NPTS * 4);
    int* invM  = (int*)take((size_t)NPTS * 4);
    int* invL  = (int*)take((size_t)NPTS * 4);
    int* vlinT = (int*)take((size_t)CELLS_T * 4);
    int* vlinM = (int*)take((size_t)MMAX * 4);
    int* vlinL = (int*)take((size_t)MMAX * 4);
    float* cntT = (float*)take((size_t)CELLS_T * 4);
    float* cntM = (float*)take((size_t)MMAX * 4);
    float* cntL = (float*)take((size_t)MMAX * 4);
    float* bn1aT = (float*)take(64 * 4); float* bn1bT = (float*)take(64 * 4);
    float* bn1aM = (float*)take(32 * 4); float* bn1bM = (float*)take(32 * 4);
    float* bn1aL = (float*)take(32 * 4); float* bn1bL = (float*)take(32 * 4);
    float* bn2aT = (float*)take(128 * 4); float* bn2bT = (float*)take(128 * 4);
    float* bn2aM = (float*)take(64 * 4);  float* bn2bM = (float*)take(64 * 4);
    float* bn2aL = (float*)take(64 * 4);  float* bn2bL = (float*)take(64 * 4);
    __bf16* hcatT = (__bf16*)take((size_t)NPTS * 128 * 2);
    __bf16* hcatM = (__bf16*)take((size_t)NPTS * 64 * 2);
    __bf16* hcatL = (__bf16*)take((size_t)NPTS * 64 * 2);
    __bf16* packT = (__bf16*)take(128 * 128 * 2);
    __bf16* packM = (__bf16*)take(64 * 64 * 2);
    __bf16* packL = (__bf16*)take(64 * 64 * 2);

    k_zero<<<2048, 256, 0, stream>>>((uint4*)zbase, zbytes / 16);

    Scale sT = {0.4f, 0.4f, 4.0f, 176, 200, 1, CELLS_T,
                countT, prefT, sumxyzT, invT, vlinT, cntT,
                bn1sT, bn1qT, bn1aT, bn1bT, bn2sT, bn2qT, bn2aT, bn2bT,
                voxsum1T, voxsum2T, hcatT, packT,
                (const float*)d_in[2], (const float*)d_in[3], (const float*)d_in[4],
                (const float*)d_in[5], (const float*)d_in[6], (const float*)d_in[7],
                Mc + 0};
    Scale sM = {0.2f, 0.2f, 2.0f, 352, 400, 2, CELLS_M,
                countM, prefM, sumxyzM, invM, vlinM, cntM,
                bn1sM, bn1qM, bn1aM, bn1bM, bn2sM, bn2qM, bn2aM, bn2bM,
                voxsum1M, voxsum2M, hcatM, packM,
                (const float*)d_in[8], (const float*)d_in[9], (const float*)d_in[10],
                (const float*)d_in[11], (const float*)d_in[12], (const float*)d_in[13],
                Mc + 1};
    Scale sL = {0.1f, 0.1f, 1.0f, 704, 800, 4, CELLS_L,
                countL, prefL, sumxyzL, invL, vlinL, cntL,
                bn1sL, bn1qL, bn1aL, bn1bL, bn2sL, bn2qL, bn2aL, bn2bL,
                voxsum1L, voxsum2L, hcatL, packL,
                (const float*)d_in[14], (const float*)d_in[15], (const float*)d_in[16],
                (const float*)d_in[17], (const float*)d_in[18], (const float*)d_in[19],
                Mc + 2};

    run_scale<64>(sT, pts, bsum, stream);
    run_scale<32>(sM, pts, bsum, stream);
    run_scale<32>(sL, pts, bsum, stream);

    float* out = (float*)d_out;
    // vf: N x 128
    k_vf<<<(NPTS * 128 + 255) / 256, 256, 0, stream>>>(voxsum2T, cntT, Mc + 0, out, 128, NPTS);
    // merge med/low voxel means into top voxels
    int mb = (MMAX + 255) / 256;
    k_merge<<<mb, 256, 0, stream>>>(vlinM, cntM, voxsum2M, Mc + 1, 352, 400, 2, 2,
                                    vlinT, Mc + 0, msum, mcnt, 64);
    k_merge<<<mb, 256, 0, stream>>>(vlinL, cntL, voxsum2L, Mc + 2, 704, 800, 4, 4,
                                    vlinT, Mc + 0, lsum, lcnt, 64);
    k_mergedfinal<<<(NPTS * 128 + 255) / 256, 256, 0, stream>>>(msum, mcnt, lsum, lcnt,
                                                                Mc + 0, out + (size_t)NPTS * 128, NPTS);
    k_coors<<<(NPTS + 255) / 256, 256, 0, stream>>>(vlinT, Mc + 0,
                                                    out + (size_t)NPTS * 256, NPTS);
}
